// WindowAttention_89275190215403
// MI455X (gfx1250) — matrix operand order repaired and checked by reading, not compiled, not run on hardware
//
#include <hip/hip_runtime.h>
#include <hip/hip_bf16.h>

typedef _Float16 f16;
typedef __attribute__((ext_vector_type(16))) _Float16 v16h;
typedef __attribute__((ext_vector_type(8)))  _Float16 h8;
typedef __attribute__((ext_vector_type(4)))  _Float16 h4;
typedef __attribute__((ext_vector_type(8)))  float    v8f;

#define NTOK   64
#define DIM    256
#define HEADS  8
#define HD     32
#define QSCALE 0.17677669529663687f  // 1/sqrt(32)

// ---------------- WMMA fragment helpers (contiguous-K layouts, wave32) ----------

// A-matrix 16x32 f16 (M x K), row-major source with leading dim `ld` (halves).
// lane 0-15:  M=lane,    K = 0..15 ; lane 16-31: M=lane-16, K = 16..31
__device__ __forceinline__ v16h frag_A(const f16* __restrict__ base, int ld) {
  const int lane = threadIdx.x & 31;
  const f16* p = base + (lane & 15) * ld + ((lane >> 4) << 4);
  h8 lo = *reinterpret_cast<const h8*>(p);        // K = kb .. kb+7
  h8 hi = *reinterpret_cast<const h8*>(p + 8);    // K = kb+8 .. kb+15
  v16h r;
#pragma unroll
  for (int i = 0; i < 8; ++i) { r[i] = lo[i]; r[i + 8] = hi[i]; }
  return r;
}

// B-matrix 32x16 f16 (K x N) where element B[k][n] lives at base[n*ld + k]
// (contiguous along K). lane 0-15: N=lane, K=0..15 ; lane 16-31: N=lane-16, K=16..31
__device__ __forceinline__ v16h frag_Bt(const f16* __restrict__ base, int ld) {
  const int lane = threadIdx.x & 31;
  return *reinterpret_cast<const v16h*>(base + (lane & 15) * ld + ((lane >> 4) << 4));
}

__device__ __forceinline__ v8f wmma_f16(v16h a, v16h b, v8f c) {
  return __builtin_amdgcn_wmma_f32_16x16x32_f16(
      /*neg_a=*/false, a, /*neg_b=*/false, b,
      /*c_mod=*/(short)0, c, /*reuse_a=*/false, /*reuse_b=*/false);
}

__device__ __forceinline__ v8f zero8() {
  v8f z = {0.f, 0.f, 0.f, 0.f, 0.f, 0.f, 0.f, 0.f};
  return z;
}

// ---------------- prep kernels ------------------------------------------------

__global__ void cvt_weights_kernel(const float* __restrict__ qkv_w,
                                   const float* __restrict__ proj_w,
                                   f16* __restrict__ wq, f16* __restrict__ wp) {
  int i = blockIdx.x * blockDim.x + threadIdx.x;
  if (i < 768 * 256) wq[i] = (f16)qkv_w[i];
  int j = i - 768 * 256;
  if (j >= 0 && j < 256 * 256) wp[j] = (f16)proj_w[j];
}

// bias[h][q][k] = rpb_table[rel_index[q][k]][h]   -> [8][64][64] f32
__global__ void build_bias_kernel(const float* __restrict__ rpb,
                                  const int* __restrict__ rel,
                                  float* __restrict__ bias) {
  int i = blockIdx.x * blockDim.x + threadIdx.x;  // 0..4095 (q*64+k)
  if (i < NTOK * NTOK) {
    int idx = rel[i];
#pragma unroll
    for (int h = 0; h < HEADS; ++h) bias[h * NTOK * NTOK + i] = rpb[idx * HEADS + h];
  }
}

// ---------------- main kernel: one workgroup per window -----------------------

__global__ void __launch_bounds__(256, 1)
swin_window_attn_kernel(const float* __restrict__ x,
                        const float* __restrict__ qkv_b,
                        const float* __restrict__ proj_b,
                        const f16*  __restrict__ wq,     // [768][256] f16 (L2-resident)
                        const f16*  __restrict__ wp,     // [256][256] f16
                        const float* __restrict__ bias,  // [8][64][64] f32
                        float* __restrict__ y) {
  extern __shared__ __align__(32) char smem[];
  f16* Qh = reinterpret_cast<f16*>(smem);   // [64][256]  32KB (pre-scaled Q)
  f16* Kh = Qh + NTOK * DIM;                // [64][256]  32KB
  f16* Vt = Kh + NTOK * DIM;                // [256][64]  32KB (V transposed)
  f16* Sb = Vt + DIM * NTOK;                // [8][64][64] 64KB scores/probs
  f16* Xh = Sb;                             // [64][256]  overlaps Sb (dead after QKV)
  f16* Oh = Qh;                             // [64][256]  overlaps Q (dead after attn)

  const int b    = blockIdx.x;
  const int tid  = threadIdx.x;
  const int lane = tid & 31;
  const int wu   = __builtin_amdgcn_readfirstlane(tid >> 5);  // uniform wave id (SGPR)
  const int n    = lane & 15;           // fragment column
  const int mo   = (lane >> 4) << 3;    // row offset 0 or 8 for C/D frags
  // Phases B/E work split: each wave owns TWO row-tiles (mt2, mt2+1) and a
  // quarter of the N-tiles -> each weight tile is read by only 2 waves, and
  // every B-fragment load feeds two WMMAs.
  const int mt2  = (wu & 1) << 1;       // 0 or 2 : first of this wave's two M tiles
  const int nt0  = wu >> 1;             // N-tile start 0..3, step 4

  // ---- Phase A: stage X as f16 in LDS --------------------------------------
  const float* xg = x + (size_t)b * NTOK * DIM;
  for (int i = tid; i < NTOK * DIM / 4; i += 256) {
    float4 v = reinterpret_cast<const float4*>(xg)[i];
    h4 hv; hv[0] = (f16)v.x; hv[1] = (f16)v.y; hv[2] = (f16)v.z; hv[3] = (f16)v.w;
    reinterpret_cast<h4*>(Xh)[i] = hv;
  }
  __syncthreads();

  // ---- Phase B: QKV = Xh @ qkv_w^T + qkv_b ---------------------------------
  // A-fragments pinned in registers; B-fragments software-pipelined (rotating
  // double buffer) so each WMMA pair overlaps the next fragment's L2 load.
  {
    v16h afr[2][8];
#pragma unroll
    for (int mm = 0; mm < 2; ++mm)
#pragma unroll
      for (int kk = 0; kk < 8; ++kk)
        afr[mm][kk] = frag_A(Xh + (mt2 + mm) * 16 * DIM + kk * 32, DIM);

    v16h bb = frag_Bt(wq + nt0 * 16 * DIM, DIM);     // preload first fragment
    for (int nt = nt0; nt < 48; nt += 4) {           // 12 N-tiles per wave (uniform)
      const int ntn = (nt + 4 < 48) ? (nt + 4) : nt; // next tile (clamped)
      // pull next 8KB weight tile toward the WGP while this one computes
      __builtin_prefetch(wq + ntn * 16 * DIM + lane * 128, 0, 3);
      v8f acc[2] = {zero8(), zero8()};
#pragma unroll
      for (int kk = 0; kk < 8; ++kk) {
        v16h bcur = bb;
        const f16* nb = (kk < 7) ? (wq + nt * 16 * DIM + (kk + 1) * 32)
                                 : (wq + ntn * 16 * DIM);
        bb = frag_Bt(nb, DIM);                       // issue load for next step
        acc[0] = wmma_f16(afr[0][kk], bcur, acc[0]);
        acc[1] = wmma_f16(afr[1][kk], bcur, acc[1]);
      }
      const int j = nt * 16 + n;            // same feature for all 8 rows of a lane
      const float bv = qkv_b[j];
      if (nt < 16) {                        // Q: scale now, row-major
#pragma unroll
        for (int mm = 0; mm < 2; ++mm)
#pragma unroll
          for (int r = 0; r < 8; ++r)
            Qh[((mt2 + mm) * 16 + mo + r) * DIM + j] =
                (f16)((acc[mm][r] + bv) * QSCALE);
      } else if (nt < 32) {                 // K: row-major
#pragma unroll
        for (int mm = 0; mm < 2; ++mm)
#pragma unroll
          for (int r = 0; r < 8; ++r)
            Kh[((mt2 + mm) * 16 + mo + r) * DIM + (j - 256)] = (f16)(acc[mm][r] + bv);
      } else {                              // V: transposed -> contiguous b128 stores
#pragma unroll
        for (int mm = 0; mm < 2; ++mm) {
          h8 hv;
#pragma unroll
          for (int r = 0; r < 8; ++r) hv[r] = (f16)(acc[mm][r] + bv);
          *reinterpret_cast<h8*>(Vt + (j - 512) * NTOK + (mt2 + mm) * 16 + mo) = hv;
        }
      }
    }
  }
  __syncthreads();

  // ---- Phase C: per-head scores S = Q_h @ K_h^T (wave wu == head) ----------
  {
    const int h = wu;
    v16h qa[4];
#pragma unroll
    for (int qt = 0; qt < 4; ++qt)
      qa[qt] = frag_A(Qh + qt * 16 * DIM + h * HD, DIM);   // 16x32, K = head dim
#pragma unroll
    for (int kt = 0; kt < 4; ++kt) {
      v16h kb = frag_Bt(Kh + kt * 16 * DIM + h * HD, DIM); // K[n][k] contiguous
#pragma unroll
      for (int qt = 0; qt < 4; ++qt) {
        v8f acc = wmma_f16(qa[qt], kb, zero8());
#pragma unroll
        for (int r = 0; r < 8; ++r)
          Sb[h * 4096 + (qt * 16 + mo + r) * 64 + kt * 16 + n] = (f16)acc[r];
      }
    }
  }
  // intra-wave only from here: this wave wrote S for its own head (DS in-order)

  // ---- softmax over k, with relative-position bias (f32 math) ---------------
  {
    const int h = wu;
#pragma unroll
    for (int rr = 0; rr < 2; ++rr) {
      const int q = lane + rr * 32;
      f16* srow = Sb + h * 4096 + q * 64;
      const float* brow = bias + h * 4096 + q * 64;
      float vbuf[64];
      float mx = -1e30f;
#pragma unroll
      for (int k = 0; k < 64; ++k) {
        vbuf[k] = (float)srow[k] + brow[k];
        mx = fmaxf(mx, vbuf[k]);
      }
      float s = 0.f;
#pragma unroll
      for (int k = 0; k < 64; ++k) { vbuf[k] = __expf(vbuf[k] - mx); s += vbuf[k]; }
      const float inv = 1.f / s;
#pragma unroll
      for (int k = 0; k < 64; ++k) srow[k] = (f16)(vbuf[k] * inv);
    }
  }

  // ---- Phase D: O_h = P_h @ V_h  (K=64 over tokens) -------------------------
  {
    const int h = wu;
    v16h vb[2][2];
#pragma unroll
    for (int nt = 0; nt < 2; ++nt)
#pragma unroll
      for (int kk = 0; kk < 2; ++kk)
        vb[nt][kk] = frag_Bt(Vt + (h * HD + nt * 16) * NTOK + kk * 32, NTOK);
#pragma unroll
    for (int qt = 0; qt < 4; ++qt) {
      v16h pa0 = frag_A(Sb + h * 4096 + qt * 16 * 64 + 0,  64);
      v16h pa1 = frag_A(Sb + h * 4096 + qt * 16 * 64 + 32, 64);
#pragma unroll
      for (int nt = 0; nt < 2; ++nt) {
        v8f acc = wmma_f16(pa0, vb[nt][0], zero8());
        acc = wmma_f16(pa1, vb[nt][1], acc);
#pragma unroll
        for (int r = 0; r < 8; ++r)
          Oh[(qt * 16 + mo + r) * DIM + h * HD + nt * 16 + n] = (f16)acc[r];
      }
    }
  }
  __syncthreads();

  // ---- Phase E: Y = Oh @ proj_w^T + proj_b ----------------------------------
  {
    v16h ofr[2][8];
#pragma unroll
    for (int mm = 0; mm < 2; ++mm)
#pragma unroll
      for (int kk = 0; kk < 8; ++kk)
        ofr[mm][kk] = frag_A(Oh + (mt2 + mm) * 16 * DIM + kk * 32, DIM);

    float* yg = y + (size_t)b * NTOK * DIM;
    v16h bb = frag_Bt(wp + nt0 * 16 * DIM, DIM);     // preload first fragment
    for (int nt = nt0; nt < 16; nt += 4) {           // 4 N-tiles per wave (uniform)
      const int ntn = (nt + 4 < 16) ? (nt + 4) : nt;
      __builtin_prefetch(wp + ntn * 16 * DIM + lane * 128, 0, 3);
      v8f acc[2] = {zero8(), zero8()};
#pragma unroll
      for (int kk = 0; kk < 8; ++kk) {
        v16h bcur = bb;
        const f16* nb = (kk < 7) ? (wp + nt * 16 * DIM + (kk + 1) * 32)
                                 : (wp + ntn * 16 * DIM);
        bb = frag_Bt(nb, DIM);
        acc[0] = wmma_f16(ofr[0][kk], bcur, acc[0]);
        acc[1] = wmma_f16(ofr[1][kk], bcur, acc[1]);
      }
      const float bv = proj_b[nt * 16 + n];
#pragma unroll
      for (int mm = 0; mm < 2; ++mm)
#pragma unroll
        for (int r = 0; r < 8; ++r)
          yg[((mt2 + mm) * 16 + mo + r) * DIM + nt * 16 + n] = acc[mm][r] + bv;
    }
  }
}

// ---------------- host launcher ----------------------------------------------

extern "C" void kernel_launch(void* const* d_in, const int* in_sizes, int n_in,
                              void* d_out, int out_size, void* d_ws, size_t ws_size,
                              hipStream_t stream) {
  const float* x      = (const float*)d_in[0];
  const float* qkv_w  = (const float*)d_in[1];
  const float* qkv_b  = (const float*)d_in[2];
  const float* proj_w = (const float*)d_in[3];
  const float* proj_b = (const float*)d_in[4];
  const float* rpb    = (const float*)d_in[5];
  const int*   rel    = (const int*)d_in[6];
  float* y = (float*)d_out;

  const int B = in_sizes[0] / (NTOK * DIM);   // 3136 windows

  // workspace: f16 weights + expanded bias (all L2-resident)
  char* ws = (char*)d_ws;
  f16*   wq   = (f16*)ws;                                      // 768*256 f16 = 384KB
  f16*   wp   = (f16*)(ws + 768 * 256 * 2);                    // 256*256 f16 = 128KB
  float* bias = (float*)(ws + 768 * 256 * 2 + 256 * 256 * 2);  // 8*64*64 f32 = 128KB

  const int nconv = 768 * 256 + 256 * 256;
  cvt_weights_kernel<<<(nconv + 255) / 256, 256, 0, stream>>>(qkv_w, proj_w, wq, wp);
  build_bias_kernel<<<(NTOK * NTOK + 255) / 256, 256, 0, stream>>>(rpb, rel, bias);

  const size_t lds_bytes =
      (size_t)(3 * NTOK * DIM + HEADS * NTOK * NTOK) * sizeof(f16);  // 160KB
  (void)hipFuncSetAttribute((const void*)swin_window_attn_kernel,
                            hipFuncAttributeMaxDynamicSharedMemorySize,
                            (int)lds_bytes);
  swin_window_attn_kernel<<<B, 256, lds_bytes, stream>>>(x, qkv_b, proj_b, wq, wp,
                                                         bias, y);
}
